// TypedGraphAttention_67997922230388
// MI455X (gfx1250) — matrix operand order repaired and checked by reading, not compiled, not run on hardware
//
#include <hip/hip_runtime.h>
#include <hip/hip_bf16.h>

#define BB    4
#define NN    2048
#define DD    256
#define HH    4
#define HEADD 64

typedef __attribute__((ext_vector_type(16))) __bf16 v16bf;
typedef __attribute__((ext_vector_type(8)))  float  v8f;

static __device__ __forceinline__ v8f wmma_bf16(v16bf a, v16bf b, v8f c) {
  return __builtin_amdgcn_wmma_f32_16x16x32_bf16(false, a, false, b, (short)0, c,
                                                 false, false);
}

union FragU { uint4 q[2]; v16bf v; };

// ---- WMMA fragment loaders (ISA 7.12.2 layouts, wave32) -------------------
// A: 16x32 bf16 tile, row-major, ld = row stride (elements, mult of 8).
static __device__ __forceinline__ v16bf load_A_rm(const __bf16* tile, int ld) {
  int lane = threadIdx.x & 31;
  const __bf16* p = tile + (lane & 15) * ld + ((lane & 16) ? 8 : 0);
  FragU u;
  u.q[0] = *(const uint4*)(p);        // K 0..7   (16B)
  u.q[1] = *(const uint4*)(p + 16);   // K 16..23 (16B)
  return u.v;
}

// A tile from fp32 source (LDS or global), converted to bf16 on the fly.
static __device__ __forceinline__ v16bf load_A_f32(const float* tile, int ld) {
  int lane = threadIdx.x & 31;
  const float* p = tile + (lane & 15) * ld + ((lane & 16) ? 8 : 0);
  float4 f0 = ((const float4*)p)[0];
  float4 f1 = ((const float4*)p)[1];
  float4 f2 = ((const float4*)(p + 16))[0];
  float4 f3 = ((const float4*)(p + 16))[1];
  v16bf a;
  a[0]  = (__bf16)f0.x; a[1]  = (__bf16)f0.y; a[2]  = (__bf16)f0.z; a[3]  = (__bf16)f0.w;
  a[4]  = (__bf16)f1.x; a[5]  = (__bf16)f1.y; a[6]  = (__bf16)f1.z; a[7]  = (__bf16)f1.w;
  a[8]  = (__bf16)f2.x; a[9]  = (__bf16)f2.y; a[10] = (__bf16)f2.z; a[11] = (__bf16)f2.w;
  a[12] = (__bf16)f3.x; a[13] = (__bf16)f3.y; a[14] = (__bf16)f3.z; a[15] = (__bf16)f3.w;
  return a;
}

// B (K x N) fragment, SOURCE stored transposed (N x K row-major): table order/lane.
static __device__ __forceinline__ v16bf load_B_tr(const __bf16* tile, int ld) {
  int lane = threadIdx.x & 31;
  const __bf16* p = tile + (lane & 15) * ld + ((lane & 16) ? 8 : 0);
  FragU u;
  u.q[0] = *(const uint4*)(p);        // K 0..7   (16B)
  u.q[1] = *(const uint4*)(p + 16);   // K 16..23 (16B)
  return u.v;
}

// ---- 1. x = nodes + node_type_embed[node_types] (fp32 + bf16 copies) ------
__global__ void tga_prep(const float* __restrict__ nodes,
                         const int*   __restrict__ ntype,
                         const float* __restrict__ nte,
                         float* __restrict__ x32, __bf16* __restrict__ x16) {
  int idx = blockIdx.x * 256 + threadIdx.x;           // B*N*D
  if (idx >= BB * NN * DD) return;
  int d  = idx & (DD - 1);
  int bn = idx >> 8;
  float v = nodes[idx] + nte[ntype[bn] * DD + d];
  x32[idx] = v;
  x16[idx] = (__bf16)v;
}

// ---- 1b. convert Wq,Wk,Wv,Wo to bf16, TRANSPOSED: wt[mat][n][k] = W[k][n] --
__global__ void tga_wcvt(const float* __restrict__ Wq, const float* __restrict__ Wk,
                         const float* __restrict__ Wv, const float* __restrict__ Wo,
                         __bf16* __restrict__ wt) {
  int idx = blockIdx.x * 256 + threadIdx.x;           // 4*D*D
  if (idx >= 4 * DD * DD) return;
  int m = idx >> 16;
  int r = idx & (DD * DD - 1);
  int n = r >> 8;
  int k = r & (DD - 1);
  float v;
  if      (m == 0) v = Wq[k * DD + n];
  else if (m == 1) v = Wk[k * DD + n];
  else if (m == 2) v = Wv[k * DD + n];
  else             v = Wo[k * DD + n];
  wt[idx] = (__bf16)v;
}

// ---- 2. QKV projection GEMMs: one wave = 16 rows x 64 cols ---------------
// Q pre-scaled by 0.125. Q,K stored [B,H,N,HEAD]; V stored TRANSPOSED [B,H,HEAD,N].
__global__ void tga_qkv(const __bf16* __restrict__ x16, const __bf16* __restrict__ wt,
                        const float* __restrict__ bq, const float* __restrict__ bk,
                        const float* __restrict__ bv,
                        __bf16* __restrict__ Q, __bf16* __restrict__ Kk,
                        __bf16* __restrict__ Vt) {
  int wg    = blockIdx.x * 8 + (threadIdx.x >> 5);    // 3 * 512 * 4 waves
  int mat   = wg >> 11;
  int rem   = wg & 2047;
  int mtile = rem >> 2;
  int ng    = rem & 3;
  int mbase = mtile * 16, nbase = ng * 64;
  const __bf16* Wt = wt + (size_t)mat * DD * DD;

  v8f c0 = {}, c1 = {}, c2 = {}, c3 = {};
#pragma unroll
  for (int k0 = 0; k0 < DD; k0 += 32) {
    v16bf a  = load_A_rm(x16 + (size_t)mbase * DD + k0, DD);
    c0 = wmma_bf16(a, load_B_tr(Wt + (size_t)(nbase +  0) * DD + k0, DD), c0);
    c1 = wmma_bf16(a, load_B_tr(Wt + (size_t)(nbase + 16) * DD + k0, DD), c1);
    c2 = wmma_bf16(a, load_B_tr(Wt + (size_t)(nbase + 32) * DD + k0, DD), c2);
    c3 = wmma_bf16(a, load_B_tr(Wt + (size_t)(nbase + 48) * DD + k0, DD), c3);
  }

  int lane  = threadIdx.x & 31;
  int moff  = mbase + ((lane & 16) ? 8 : 0);
  const float* bias = (mat == 0) ? bq : ((mat == 1) ? bk : bv);
  float scale = (mat == 0) ? 0.125f : 1.0f;
  v8f acc[4] = {c0, c1, c2, c3};
#pragma unroll
  for (int j = 0; j < 4; ++j) {
    int n  = nbase + j * 16 + (lane & 15);
    int h  = n >> 6, dh = n & 63;
    float bsn = bias[n];
#pragma unroll
    for (int r = 0; r < 8; ++r) {
      int m = moff + r;
      int b_ = m >> 11, nrow = m & (NN - 1);
      __bf16 val = (__bf16)((acc[j][r] + bsn) * scale);
      if (mat == 0)
        Q[(((size_t)(b_ * HH + h) * NN) + nrow) * HEADD + dh] = val;
      else if (mat == 1)
        Kk[(((size_t)(b_ * HH + h) * NN) + nrow) * HEADD + dh] = val;
      else
        Vt[(((size_t)(b_ * HH + h) * HEADD) + dh) * NN + nrow] = val;
    }
  }
}

// ---- 3. FUSED attention: scores + biases + softmax + attn@V ---------------
// One workgroup (8 waves, 128KB LDS) = (b,h, 16 query rows).
//   phase 1: wave w computes score cols [256w, 256w+256) -> LDS (16 x 2048 fp32)
//   phase 2: wave w softmaxes rows {2w, 2w+1} in LDS, writes normalized attn out
//   phase 3: wave w multiplies its K-slice of LDS-attn against V^T (WMMA),
//            partials reduced across waves through LDS -> ctx [B,N,D] bf16
__global__ void tga_attn(const __bf16* __restrict__ Q, const __bf16* __restrict__ Kk,
                         const __bf16* __restrict__ Vt,
                         const int* __restrict__ etypes, const float* __restrict__ eemb,
                         const int* __restrict__ ntypes, const float* __restrict__ adj,
                         float* __restrict__ attn, __bf16* __restrict__ ctx) {
  extern __shared__ float sc[];                       // 16 * 2048 fp32 = 128KB
  int bh    = blockIdx.x >> 7;                        // B*H*128 blocks
  int mtile = blockIdx.x & 127;
  int b = bh >> 2, h = bh & 3;
  int mbase = mtile * 16;
  int w    = threadIdx.x >> 5;
  int lane = threadIdx.x & 31;

  const __bf16* Qb = Q  + (size_t)bh * NN * HEADD;
  const __bf16* Kb = Kk + (size_t)bh * NN * HEADD;
  const int* ntb = ntypes + b * NN;

  // ---- phase 1: scores + biases into LDS ----
  v16bf a0 = load_A_rm(Qb + (size_t)mbase * HEADD + 0,  HEADD);
  v16bf a1 = load_A_rm(Qb + (size_t)mbase * HEADD + 32, HEADD);

  int rloc = (lane & 16) ? 8 : 0;                     // local row base for this half
  int ty_row[8];
#pragma unroll
  for (int r = 0; r < 8; ++r) ty_row[r] = ntb[mbase + rloc + r];

  for (int nt = 0; nt < 16; ++nt) {
    int nbase = w * 256 + nt * 16;
    v8f c = {};
    c = wmma_bf16(a0, load_B_tr(Kb + (size_t)nbase * HEADD + 0,  HEADD), c);
    c = wmma_bf16(a1, load_B_tr(Kb + (size_t)nbase * HEADD + 32, HEADD), c);

    int n  = nbase + (lane & 15);
    int tn = ntb[n];
#pragma unroll
    for (int r = 0; r < 8; ++r) {
      int m = mbase + rloc + r;
      size_t e = (size_t)b * NN * NN + (size_t)m * NN + n;
      float s = c[r] + eemb[etypes[e] * HH + h];
      s += (ty_row[r] == tn) ? 0.10f : -0.05f;
      s += (1.0f - adj[e]) * -1e9f;
      sc[(rloc + r) * NN + n] = s;
    }
  }
  __syncthreads();

  // ---- phase 2: softmax rows {2w, 2w+1}; write normalized attn to global ----
#pragma unroll
  for (int rr = 0; rr < 2; ++rr) {
    int row = 2 * w + rr;
    float* p = sc + row * NN;
    float mx = -3.4e38f;
#pragma unroll
    for (int i = 0; i < 64; ++i) mx = fmaxf(mx, p[lane + i * 32]);
#pragma unroll
    for (int off = 16; off; off >>= 1) mx = fmaxf(mx, __shfl_xor(mx, off, 32));
    float sum = 0.f;
#pragma unroll
    for (int i = 0; i < 64; ++i) {
      float e_ = __expf(p[lane + i * 32] - mx);
      p[lane + i * 32] = e_;
      sum += e_;
    }
#pragma unroll
    for (int off = 16; off; off >>= 1) sum += __shfl_xor(sum, off, 32);
    float inv = 1.0f / sum;
    float4* p4 = (float4*)p;
    float4* a4 = (float4*)(attn + (size_t)bh * NN * NN + (size_t)(mbase + row) * NN);
#pragma unroll
    for (int i = 0; i < 16; ++i) {
      int c4 = lane + i * 32;
      float4 v = p4[c4];
      v.x *= inv; v.y *= inv; v.z *= inv; v.w *= inv;
      p4[c4] = v;
      a4[c4] = v;
    }
  }
  __syncthreads();

  // ---- phase 3: ctx partials over this wave's K-slice [256w, 256w+256) ----
  const __bf16* Vb = Vt + (size_t)bh * HEADD * NN;
  v8f c0 = {}, c1 = {}, c2 = {}, c3 = {};
#pragma unroll
  for (int j = 0; j < 8; ++j) {
    int k0 = w * 256 + j * 32;
    v16bf a = load_A_f32(sc + k0, NN);
    c0 = wmma_bf16(a, load_B_tr(Vb + (size_t) 0 * NN + k0, NN), c0);
    c1 = wmma_bf16(a, load_B_tr(Vb + (size_t)16 * NN + k0, NN), c1);
    c2 = wmma_bf16(a, load_B_tr(Vb + (size_t)32 * NN + k0, NN), c2);
    c3 = wmma_bf16(a, load_B_tr(Vb + (size_t)48 * NN + k0, NN), c3);
  }
  __syncthreads();                                    // done reading score LDS

  // stash partials in (reused) LDS: part[w][j*256 + r*32 + lane]
  v8f acc[4] = {c0, c1, c2, c3};
  float* part = sc;
#pragma unroll
  for (int j = 0; j < 4; ++j)
#pragma unroll
    for (int r = 0; r < 8; ++r)
      part[w * 1024 + j * 256 + r * 32 + lane] = acc[j][r];
  __syncthreads();

  // cross-wave reduction: 1024 outputs, 4 per thread
  int t = threadIdx.x;
#pragma unroll
  for (int i = 0; i < 4; ++i) {
    int idx = t + i * 256;
    float s = 0.f;
#pragma unroll
    for (int ww = 0; ww < 8; ++ww) s += part[ww * 1024 + idx];
    int j = idx >> 8, r = (idx >> 5) & 7, l = idx & 31;
    int m = mbase + ((l & 16) ? 8 : 0) + r;
    int d = h * HEADD + j * 16 + (l & 15);
    ctx[((size_t)b * NN + m) * DD + d] = (__bf16)s;
  }
}

// ---- 4. out = ctx @ Wo + bo (fp32 result); one wave = 16 x 64 -------------
__global__ void tga_oproj(const __bf16* __restrict__ ctx, const __bf16* __restrict__ Wot,
                          const float* __restrict__ bo, float* __restrict__ outp) {
  int wg    = blockIdx.x * 8 + (threadIdx.x >> 5);    // 512*4 waves
  int mtile = wg >> 2;
  int ng    = wg & 3;
  int mbase = mtile * 16, nbase = ng * 64;

  v8f c0 = {}, c1 = {}, c2 = {}, c3 = {};
#pragma unroll
  for (int k0 = 0; k0 < DD; k0 += 32) {
    v16bf a = load_A_rm(ctx + (size_t)mbase * DD + k0, DD);
    c0 = wmma_bf16(a, load_B_tr(Wot + (size_t)(nbase +  0) * DD + k0, DD), c0);
    c1 = wmma_bf16(a, load_B_tr(Wot + (size_t)(nbase + 16) * DD + k0, DD), c1);
    c2 = wmma_bf16(a, load_B_tr(Wot + (size_t)(nbase + 32) * DD + k0, DD), c2);
    c3 = wmma_bf16(a, load_B_tr(Wot + (size_t)(nbase + 48) * DD + k0, DD), c3);
  }

  int lane = threadIdx.x & 31;
  int moff = mbase + ((lane & 16) ? 8 : 0);
  v8f acc[4] = {c0, c1, c2, c3};
#pragma unroll
  for (int j = 0; j < 4; ++j) {
    int n = nbase + j * 16 + (lane & 15);
    float bn = bo[n];
#pragma unroll
    for (int r = 0; r < 8; ++r)
      outp[(size_t)(moff + r) * DD + n] = acc[j][r] + bn;
  }
}

// ---- 5. y = LayerNorm(x + out) --------------------------------------------
__global__ void tga_ln(const float* __restrict__ x32, const float* __restrict__ outp,
                       const float* __restrict__ g, const float* __restrict__ bta,
                       float* __restrict__ y) {
  size_t base = (size_t)blockIdx.x * DD;
  int t = threadIdx.x;                                 // 256 threads = D
  float v = x32[base + t] + outp[base + t];
  float s = v, sq = v * v;
#pragma unroll
  for (int off = 16; off; off >>= 1) {
    s  += __shfl_xor(s, off, 32);
    sq += __shfl_xor(sq, off, 32);
  }
  __shared__ float ss[8], ssq[8];
  int wv = t >> 5;
  if ((t & 31) == 0) { ss[wv] = s; ssq[wv] = sq; }
  __syncthreads();
  float S = 0.f, SQ = 0.f;
#pragma unroll
  for (int i = 0; i < 8; ++i) { S += ss[i]; SQ += ssq[i]; }
  float mu  = S * (1.0f / DD);
  float var = SQ * (1.0f / DD) - mu * mu;
  y[base + t] = (v - mu) * rsqrtf(var + 1e-5f) * g[t] + bta[t];
}

extern "C" void kernel_launch(void* const* d_in, const int* in_sizes, int n_in,
                              void* d_out, int out_size, void* d_ws, size_t ws_size,
                              hipStream_t stream) {
  const float* nodes  = (const float*)d_in[0];
  const int*   ntypes = (const int*)  d_in[1];
  const int*   etypes = (const int*)  d_in[2];
  const float* adj    = (const float*)d_in[3];
  const float* nte    = (const float*)d_in[4];
  const float* eemb   = (const float*)d_in[5];
  const float* Wq = (const float*)d_in[6];  const float* bq = (const float*)d_in[7];
  const float* Wk = (const float*)d_in[8];  const float* bk = (const float*)d_in[9];
  const float* Wv = (const float*)d_in[10]; const float* bv = (const float*)d_in[11];
  const float* Wo = (const float*)d_in[12]; const float* bo = (const float*)d_in[13];
  const float* lng = (const float*)d_in[14]; const float* lnb = (const float*)d_in[15];

  float* y    = (float*)d_out;                          // [B,N,D]
  float* attn = y + (size_t)BB * NN * DD;               // [B,H,N,N]

  // workspace carve-up (~36.5 MiB)
  char* ws = (char*)d_ws;
  float*  x32  = (float*)ws;  ws += (size_t)BB * NN * DD * 4;
  float*  outp = (float*)ws;  ws += (size_t)BB * NN * DD * 4;
  __bf16* x16  = (__bf16*)ws; ws += (size_t)BB * NN * DD * 2;
  __bf16* wt   = (__bf16*)ws; ws += (size_t)4 * DD * DD * 2;
  __bf16* Qb   = (__bf16*)ws; ws += (size_t)BB * NN * DD * 2;
  __bf16* Kb   = (__bf16*)ws; ws += (size_t)BB * NN * DD * 2;
  __bf16* Vt   = (__bf16*)ws; ws += (size_t)BB * NN * DD * 2;
  __bf16* ctx  = (__bf16*)ws; ws += (size_t)BB * NN * DD * 2;

  // allow 128KB dynamic LDS for the fused attention kernel (CDNA5: 320KB/WGP)
  const int ATTN_LDS = 16 * NN * (int)sizeof(float);   // 131072
  (void)hipFuncSetAttribute((const void*)tga_attn,
                            hipFuncAttributeMaxDynamicSharedMemorySize, ATTN_LDS);

  tga_prep<<<(BB * NN * DD) / 256, 256, 0, stream>>>(nodes, ntypes, nte, x32, x16);
  tga_wcvt<<<(4 * DD * DD) / 256, 256, 0, stream>>>(Wq, Wk, Wv, Wo, wt);
  tga_qkv<<<(3 * 512 * 4) / 8, 256, 0, stream>>>(x16, wt, bq, bk, bv, Qb, Kb, Vt);
  tga_attn<<<BB * HH * 128, 256, ATTN_LDS, stream>>>(Qb, Kb, Vt, etypes, eemb,
                                                     ntypes, adj, attn, ctx);
  tga_oproj<<<(512 * 4) / 8, 256, 0, stream>>>(ctx, wt + (size_t)3 * DD * DD, bo, outp);
  tga_ln<<<BB * NN, 256, 0, stream>>>(x32, outp, lng, lnb, y);
}
